// STUBlock_27573690040575
// MI455X (gfx1250) — matrix operand order repaired and checked by reading, not compiled, not run on hardware
//
#include <hip/hip_runtime.h>
#include <cmath>

// ---------------------------------------------------------------------------
// STU block for MI455X (gfx1250, wave32, WMMA bf16 / f32-accumulate).
// Causal spectral conv = lower-triangular Toeplitz matmul. Toeplitz A-tiles
// are precomputed per (filter, t0-s0) into a 3MB L2-resident table, so the
// conv inner loop is barrier-free: b128 loads + 4x v_wmma per A-fragment.
// All B operands are pre-transposed K-contiguous bf16.
// ---------------------------------------------------------------------------

constexpr int B_ = 2;
constexpr int L_ = 2048;
constexpr int D_ = 768;
constexpr int K_ = 24;
constexpr int H_ = 3072;         // SCALE * D
constexpr int KD_ = K_ * D_;     // 18432
constexpr int NDIFF_ = 128;      // distinct (t0-s0)/16 values
constexpr float EPS_ = 1e-5f;

typedef __attribute__((ext_vector_type(16))) __bf16 v16bf;
typedef __attribute__((ext_vector_type(8)))  float  v8f;

static __device__ inline v8f zero8() {
  v8f z = {0.f, 0.f, 0.f, 0.f, 0.f, 0.f, 0.f, 0.f};
  return z;
}

static __device__ inline v8f wmma_bf16(v16bf a, v16bf b, v8f c) {
  // (neg_a, A, neg_b, B, c_mod, C, reuse_a, reuse_b)
  return __builtin_amdgcn_wmma_f32_16x16x32_bf16(false, a, false, b, (short)0, c,
                                                 false, false);
}

// A fragment from a row-major [16 x ld] bf16 tile (ld multiple of 8).
// ISA 7.12.2: lane m=lane&15; K halves {kb..kb+7} and {16+kb..16+kb+7}, kb=0|8.
static __device__ inline v16bf load_a_frag_rm(const __bf16* t, int ld) {
  const int lane = threadIdx.x & 31;
  const int m  = lane & 15;
  const int kb = (lane & 16) >> 1;           // 0 or 8
  v16bf f;
  uint4* q = reinterpret_cast<uint4*>(&f);
  q[0] = *reinterpret_cast<const uint4*>(t + m * ld + kb);
  q[1] = *reinterpret_cast<const uint4*>(t + m * ld + 16 + kb);
  return f;
}

// B fragment from K-contiguous (transposed) storage: row n holds consecutive K.
static __device__ inline v16bf load_b_frag_kc(const __bf16* t, size_t ldk) {
  const int lane = threadIdx.x & 31;
  const int n  = lane & 15;
  const int kb = (lane & 16) >> 1;           // 0 or 8
  v16bf f;
  uint4* q = reinterpret_cast<uint4*>(&f);
  const __bf16* p = t + (size_t)n * ldk + kb;
  q[0] = *reinterpret_cast<const uint4*>(p);
  q[1] = *reinterpret_cast<const uint4*>(p + 16);
  return f;
}

// --------------------------- LayerNorm -> bf16 -----------------------------
// transposed==1: write [B][D][L] (time-major, for conv B operand)
__global__ void ln_bf16_kernel(const float* __restrict__ x,
                               const float* __restrict__ w,
                               const float* __restrict__ b,
                               __bf16* __restrict__ out, int transposed) {
  __shared__ float red[256];
  const size_t row = blockIdx.x;
  const float* xr = x + row * D_;
  const int tid = threadIdx.x;
  float xv[3];
  float s = 0.f;
#pragma unroll
  for (int i = 0; i < 3; ++i) { xv[i] = xr[tid + 256 * i]; s += xv[i]; }
  red[tid] = s;
  __syncthreads();
  for (int off = 128; off > 0; off >>= 1) {
    if (tid < off) red[tid] += red[tid + off];
    __syncthreads();
  }
  const float mu = red[0] * (1.f / D_);
  __syncthreads();
  float vs = 0.f;
#pragma unroll
  for (int i = 0; i < 3; ++i) { const float d = xv[i] - mu; vs += d * d; }
  red[tid] = vs;
  __syncthreads();
  for (int off = 128; off > 0; off >>= 1) {
    if (tid < off) red[tid] += red[tid + off];
    __syncthreads();
  }
  const float rstd = rsqrtf(red[0] * (1.f / D_) + EPS_);
  const int bb = (int)(row / L_);
  const int l  = (int)(row % L_);
#pragma unroll
  for (int i = 0; i < 3; ++i) {
    const int c = tid + 256 * i;
    const __bf16 val = (__bf16)((xv[i] - mu) * rstd * w[c] + b[c]);
    if (transposed)
      out[((size_t)bb * D_ + c) * L_ + l] = val;
    else
      out[row * D_ + c] = val;
  }
}

// ---------------- zero-padded, eig_vals^0.25-scaled filters ----------------
__global__ void prep_phi_kernel(const float* __restrict__ eig_vals,
                                const float* __restrict__ eig_vecs,
                                __bf16* __restrict__ phi_ext) {
  const int k = blockIdx.x;
  const float sc = powf(eig_vals[k], 0.25f);
  for (int j = threadIdx.x; j < 2 * L_; j += 256) {
    float val = 0.f;
    const int t = j - (L_ - 1);
    if (t >= 0 && t < L_) val = eig_vecs[(size_t)t * K_ + k] * sc;
    phi_ext[(size_t)k * 2 * L_ + j] = (__bf16)val;
  }
}

// ------- precompute Toeplitz A-tiles: a_pre[k][diff/16][16x32] -------------
// A[m][kk] = phi_k[diff + m - kk]; diff = 16*di, di in [0,128)
__global__ void prep_atiles_kernel(const __bf16* __restrict__ phi_ext,
                                   __bf16* __restrict__ a_pre) {
  const int k  = blockIdx.x >> 7;
  const int di = blockIdx.x & (NDIFF_ - 1);
  const int diff = di * 16;
  const __bf16* phik = phi_ext + (size_t)k * (2 * L_);
  __bf16* dst = a_pre + (size_t)blockIdx.x * 512;
  for (int i = threadIdx.x; i < 512; i += 256) {
    const int m = i >> 5, kk = i & 31;
    dst[i] = phik[diff + m - kk + (L_ - 1)];
  }
}

// --------------- fp32 [R][C] -> transposed bf16 [C][R] ---------------------
__global__ void t_convert_kernel(const float* __restrict__ in,
                                 __bf16* __restrict__ out, int R, int C) {
  const size_t i = (size_t)blockIdx.x * blockDim.x + threadIdx.x;
  if (i < (size_t)R * C) {
    const size_t r = i / C, c = i % C;
    out[c * (size_t)R + r] = (__bf16)in[i];
  }
}

__global__ void copy_kernel(const float* __restrict__ src,
                            float* __restrict__ dst, int n) {
  const int i = blockIdx.x * blockDim.x + threadIdx.x;
  if (i < n) dst[i] = src[i];
}

// --------- causal Toeplitz conv for one filter k: Y_k = T_k @ U ------------
// grid (D/256, L/16, B), 128 threads; wave owns 64 d-cols (4 WMMA tiles).
// Zero LDS, zero barriers: A-fragments come from the precomputed tile table,
// B-fragments stream from time-major ut; both are pairs of b128 loads.
__global__ void conv_wmma_kernel(const __bf16* __restrict__ a_pre, // [K][128][512]
                                 const __bf16* __restrict__ ut,    // [B][D][L]
                                 __bf16* __restrict__ yk,          // [B*L][D]
                                 int kidx) {
  const int tid = threadIdx.x;
  const int wave = tid >> 5;
  const int lane = tid & 31;
  const int n  = lane & 15;
  const int mb = (lane >> 4) << 3;
  const int d0 = blockIdx.x * 256 + wave * 64;
  const int t0 = blockIdx.y * 16;
  const int b  = blockIdx.z;
  const __bf16* atab = a_pre + (size_t)kidx * NDIFF_ * 512;

  v8f c[4];
#pragma unroll
  for (int j = 0; j < 4; ++j) c[j] = zero8();

  for (int s0 = 0; s0 <= t0 + 15; s0 += 32) {        // causal K-loop
    const int di = (t0 - s0) >> 4;                   // in [0, 128)
    const v16bf a = load_a_frag_rm(atab + (size_t)di * 512, 32);
#pragma unroll
    for (int j = 0; j < 4; ++j) {
      const v16bf bb =
          load_b_frag_kc(ut + ((size_t)b * D_ + d0 + j * 16) * L_ + s0, L_);
      c[j] = wmma_bf16(a, bb, c[j]);
    }
  }
#pragma unroll
  for (int j = 0; j < 4; ++j)
#pragma unroll
    for (int r = 0; r < 8; ++r)
      yk[((size_t)b * L_ + t0 + mb + r) * D_ + d0 + j * 16 + n] = (__bf16)c[j][r];
}

// ------------- acc += Y_k @ m_phi[k*D:(k+1)*D, :]  (LDS-free) --------------
__global__ void gemm_acc_kernel(const __bf16* __restrict__ yk,      // [B*L][D]
                                const __bf16* __restrict__ m_phi_t, // [D][K*D]
                                float* __restrict__ acc, int kidx) {
  const int tid = threadIdx.x;
  const int wave = tid >> 5;
  const int lane = tid & 31;
  const int n  = lane & 15;
  const int mb = (lane >> 4) << 3;
  const int d0 = blockIdx.x * 256 + wave * 64;
  const size_t row0 = (size_t)blockIdx.y * 16;

  v8f c[4];
#pragma unroll
  for (int j = 0; j < 4; ++j) c[j] = zero8();

  for (int k0 = 0; k0 < D_; k0 += 32) {
    const v16bf a = load_a_frag_rm(yk + row0 * D_ + k0, D_);
#pragma unroll
    for (int j = 0; j < 4; ++j) {
      const v16bf bb = load_b_frag_kc(
          m_phi_t + (size_t)(d0 + j * 16) * KD_ + (size_t)kidx * D_ + k0,
          (size_t)KD_);
      c[j] = wmma_bf16(a, bb, c[j]);
    }
  }
#pragma unroll
  for (int j = 0; j < 4; ++j)
#pragma unroll
    for (int r = 0; r < 8; ++r) {
      float* p = acc + (row0 + mb + r) * D_ + d0 + j * 16 + n;
      *p += c[j][r];
    }
}

// ----------------- fused FFN: out = resid + proj(gelu(fc(v))) --------------
__global__ void ffn_kernel(const __bf16* __restrict__ v,        // [B*L][D]
                           const __bf16* __restrict__ fc_wt,    // [H][D]
                           const float* __restrict__ fc_b,      // [H]
                           const __bf16* __restrict__ proj_wt,  // [D][H]
                           const float* __restrict__ proj_b,    // [D]
                           const float* __restrict__ resid,     // [B*L][D]
                           float* __restrict__ out) {           // [B*L][D]
  __shared__ alignas(16) __bf16 sH[16 * D_];   // 24 KB hidden chunk
  const int tid = threadIdx.x;
  const int wave = tid >> 5;
  const int lane = tid & 31;
  const int n  = lane & 15;
  const int mb = (lane >> 4) << 3;
  const size_t row0 = (size_t)blockIdx.x * 16;

  v8f c2[3];
#pragma unroll
  for (int t = 0; t < 3; ++t) c2[t] = zero8();

  for (int c0 = 0; c0 < H_; c0 += D_) {        // 4 hidden chunks of 768
    // GEMM1 + bias + exact GELU -> sH (wave owns 64 cols per pass)
    for (int nc = 0; nc < D_; nc += 256) {
      const int n0 = c0 + nc + wave * 64;
      v8f c[4];
#pragma unroll
      for (int j = 0; j < 4; ++j) c[j] = zero8();
      for (int k0 = 0; k0 < D_; k0 += 32) {
        const v16bf a = load_a_frag_rm(v + row0 * D_ + k0, D_);
        __builtin_prefetch(fc_wt + (size_t)(n0 + n) * D_ + k0 + 32, 0, 1);
#pragma unroll
        for (int j = 0; j < 4; ++j) {
          const v16bf bb = load_b_frag_kc(fc_wt + (size_t)(n0 + j * 16) * D_ + k0,
                                          (size_t)D_);
          c[j] = wmma_bf16(a, bb, c[j]);
        }
      }
#pragma unroll
      for (int j = 0; j < 4; ++j)
#pragma unroll
        for (int r = 0; r < 8; ++r) {
          const int col = n0 + j * 16 + n;
          float h = c[j][r] + fc_b[col];
          h = 0.5f * h * (1.0f + erff(h * 0.70710678118f));
          sH[(mb + r) * D_ + (col - c0)] = (__bf16)h;
        }
    }
    __syncthreads();
    // GEMM2 partial accumulation over this hidden chunk
#pragma unroll
    for (int t = 0; t < 3; ++t) {
      const int n0o = (wave + 4 * t) * 16;
      for (int k0 = 0; k0 < D_; k0 += 32) {
        const v16bf a = load_a_frag_rm(sH + k0, D_);
        const v16bf bb = load_b_frag_kc(proj_wt + (size_t)n0o * H_ + c0 + k0,
                                        (size_t)H_);
        c2[t] = wmma_bf16(a, bb, c2[t]);
      }
    }
    __syncthreads();
  }
#pragma unroll
  for (int t = 0; t < 3; ++t) {
    const int n0o = (wave + 4 * t) * 16;
#pragma unroll
    for (int r = 0; r < 8; ++r) {
      const size_t orow = row0 + mb + r;
      out[orow * D_ + n0o + n] = resid[orow * D_ + n0o + n] + c2[t][r] + proj_b[n0o + n];
    }
  }
}

// ---------------------------------------------------------------------------
extern "C" void kernel_launch(void* const* d_in, const int* in_sizes, int n_in,
                              void* d_out, int out_size, void* d_ws, size_t ws_size,
                              hipStream_t stream) {
  (void)in_sizes; (void)n_in; (void)out_size; (void)ws_size;
  const float* x        = (const float*)d_in[0];
  const float* m_phi    = (const float*)d_in[1];
  const float* eig_vals = (const float*)d_in[2];
  const float* eig_vecs = (const float*)d_in[3];
  const float* ln1_w    = (const float*)d_in[4];
  const float* ln1_b    = (const float*)d_in[5];
  const float* ln2_w    = (const float*)d_in[6];
  const float* ln2_b    = (const float*)d_in[7];
  const float* fc_w     = (const float*)d_in[8];
  const float* fc_b     = (const float*)d_in[9];
  const float* proj_w   = (const float*)d_in[10];
  const float* proj_b   = (const float*)d_in[11];
  float* out = (float*)d_out;

  char* ws = (char*)d_ws;
  size_t off = 0;
  auto carve = [&](size_t bytes) {
    char* p = ws + off;
    off += (bytes + 255) & ~(size_t)255;
    return p;
  };
  __bf16* ut      = (__bf16*)carve((size_t)B_ * D_ * L_ * 2);      // LN1, time-major
  __bf16* phi     = (__bf16*)carve((size_t)K_ * 2 * L_ * 2);       // padded filters
  __bf16* a_pre   = (__bf16*)carve((size_t)K_ * NDIFF_ * 512 * 2); // Toeplitz tiles
  __bf16* yk      = (__bf16*)carve((size_t)B_ * L_ * D_ * 2);      // per-k conv out
  float*  acc     = (float*) carve((size_t)B_ * L_ * D_ * 4);      // x + STU (fp32)
  __bf16* v2      = (__bf16*)carve((size_t)B_ * L_ * D_ * 2);      // LN2 out
  __bf16* m_phi_t = (__bf16*)carve((size_t)D_ * KD_ * 2);          // [D][K*D]
  __bf16* fc_wt   = (__bf16*)carve((size_t)H_ * D_ * 2);           // [H][D]
  __bf16* proj_wt = (__bf16*)carve((size_t)D_ * H_ * 2);           // [D][H]

  const int nElem = B_ * L_ * D_;
  ln_bf16_kernel<<<B_ * L_, 256, 0, stream>>>(x, ln1_w, ln1_b, ut, 1);
  prep_phi_kernel<<<K_, 256, 0, stream>>>(eig_vals, eig_vecs, phi);
  prep_atiles_kernel<<<K_ * NDIFF_, 256, 0, stream>>>(phi, a_pre);
  copy_kernel<<<nElem / 256, 256, 0, stream>>>(x, acc, nElem);
  t_convert_kernel<<<(KD_ * D_ + 255) / 256, 256, 0, stream>>>(m_phi, m_phi_t, KD_, D_);
  t_convert_kernel<<<(D_ * H_ + 255) / 256, 256, 0, stream>>>(fc_w, fc_wt, D_, H_);
  t_convert_kernel<<<(H_ * D_ + 255) / 256, 256, 0, stream>>>(proj_w, proj_wt, H_, D_);

  for (int k = 0; k < K_; ++k) {
    conv_wmma_kernel<<<dim3(D_ / 256, L_ / 16, B_), 128, 0, stream>>>(a_pre, ut, yk, k);
    gemm_acc_kernel<<<dim3(D_ / 256, (B_ * L_) / 16), 128, 0, stream>>>(yk, m_phi_t,
                                                                        acc, k);
  }

  ln_bf16_kernel<<<B_ * L_, 256, 0, stream>>>(acc, ln2_w, ln2_b, v2, 0);
  ffn_kernel<<<(B_ * L_) / 16, 128, 0, stream>>>(v2, fc_wt, fc_b, proj_wt, proj_b,
                                                 acc, out);
}
